// AttentionPairBias_20306605375696
// MI455X (gfx1250) — matrix operand order repaired and checked by reading, not compiled, not run on hardware
//
#include <hip/hip_runtime.h>
#include <hip/hip_bf16.h>
#include <stdint.h>

typedef _Float16 half_t;
typedef __attribute__((ext_vector_type(16))) _Float16 v16h;
typedef __attribute__((ext_vector_type(8)))  float    v8f;

namespace {
constexpr int Bc = 2;      // batch
constexpr int Nc = 512;    // sequence
constexpr int CS = 1024;   // c_s
constexpr int Hc = 16;     // heads
constexpr int Dc = 64;     // head dim
constexpr int CZ = 128;    // c_z
constexpr int BN = Bc * Nc;
}

__device__ __forceinline__ int lane_id() { return threadIdx.x & 31; }

// A-fragment element->K mapping for V_WMMA_F32_16X16X32_F16 (16x32 f16 A).
// ISA 7.12.2: lanes 0-15 M=0-15; VGPR0..3 hold K pairs 0..7 (lanes<16) or
// 8..15 (lanes>=16); VGPR4..7 hold K 16..23 / 24..31.
__device__ __forceinline__ int a_frag_k(int l, int e) {
  int j  = e >> 1;
  int kb = (l >> 4) * 8;
  return (j < 4) ? (kb + 2 * j + (e & 1))
                 : (16 + kb + 2 * (j - 4) + (e & 1));
}

// Gather an A fragment from a row-major f16 buffer (row stride ld elements).
// For a fixed lane the element->address map is two contiguous 8-half runs, so
// the compiler lowers this to two ds_load_b128 / global b128 loads.
__device__ __forceinline__ v16h load_a_f16(const half_t* base, int ld) {
  int l = lane_id();
  int m = l & 15;
  v16h a;
#pragma unroll
  for (int e = 0; e < 16; ++e) a[e] = base[m * ld + a_frag_k(l, e)];
  return a;
}

__device__ __forceinline__ v8f wmma_f16(v16h a, v16h b, v8f c) {
  return __builtin_amdgcn_wmma_f32_16x16x32_f16(
      false, a, false, b, (short)0, c, false, false);
}

// Async byte-for-byte global->LDS copy of 32B per thread (ASYNCcnt path).
// INST_OFFSET applies to both the LDS and global address (ISA §10 async ops),
// so one address pair + offset:16 moves both halves.
__device__ __forceinline__ void async_copy_b256(unsigned lds_off,
                                                const void* gsrc) {
  uint64_t ga = (uint64_t)(uintptr_t)gsrc;
  asm volatile(
      "global_load_async_to_lds_b128 %0, %1, off\n\t"
      "global_load_async_to_lds_b128 %0, %1, off offset:16"
      :
      : "v"(lds_off), "v"(ga)
      : "memory");
}

__device__ __forceinline__ void wait_asynccnt0() {
  asm volatile("s_wait_asynccnt 0" ::: "memory");
}

// ---------------------------------------------------------------------------
// Kernel 1: fused projections.  X(1024x1024) @ [wq|wk|wv|wg] (1024x4096).
// Block = 128 threads (4 waves), 64x64 output tile, K-step 32.
// ---------------------------------------------------------------------------
__global__ __launch_bounds__(128) void proj_kernel(
    const float* __restrict__ s, const float* __restrict__ k_in,
    const float* __restrict__ wq, const float* __restrict__ bq,
    const float* __restrict__ wk, const float* __restrict__ wv,
    const float* __restrict__ wg,
    half_t* __restrict__ qh, half_t* __restrict__ khT,
    half_t* __restrict__ vh, float* __restrict__ gsig) {
  __shared__ __align__(32) half_t As[64 * 32];
  __shared__ __align__(32) half_t Bs[32 * 64];

  const int m0     = blockIdx.x * 64;
  const int ngbase = blockIdx.y * 64;
  const int wsel   = ngbase >> 10;          // 0=q 1=k 2=v 3=g (uniform)
  const int ncol0  = ngbase & (CS - 1);
  const float* X = (wsel == 0 || wsel == 3) ? s : k_in;
  const float* W = (wsel == 0) ? wq : (wsel == 1) ? wk : (wsel == 2) ? wv : wg;

  const int tid = threadIdx.x, wave = tid >> 5, l = tid & 31;

  v8f acc[4];
#pragma unroll
  for (int nt = 0; nt < 4; ++nt) { v8f z = {}; acc[nt] = z; }

  for (int k0 = 0; k0 < CS; k0 += 32) {
    {  // stage A: 64 rows x 32 cols, f32 -> f16
      int r = tid >> 1, ch = (tid & 1) * 16;
      const float* src = X + (long)(m0 + r) * CS + k0 + ch;
#pragma unroll
      for (int u = 0; u < 16; ++u) As[r * 32 + ch + u] = (half_t)src[u];
    }
    {  // stage B: 32 rows x 64 cols
      int kr = tid >> 2, cq = (tid & 3) * 16;
      const float* src = W + (long)(k0 + kr) * CS + ncol0 + cq;
#pragma unroll
      for (int u = 0; u < 16; ++u) Bs[kr * 64 + cq + u] = (half_t)src[u];
    }
    __syncthreads();
    v16h a = load_a_f16(As + (wave * 16) * 32, 32);
#pragma unroll
    for (int nt = 0; nt < 4; ++nt) {
      v16h b;  // lane = N column, elements = K (table order)
#pragma unroll
      for (int e = 0; e < 16; ++e) b[e] = Bs[a_frag_k(l, e) * 64 + nt * 16 + (l & 15)];
      acc[nt] = wmma_f16(a, b, acc[nt]);
    }
    __syncthreads();
  }

#pragma unroll
  for (int nt = 0; nt < 4; ++nt) {
#pragma unroll
    for (int r = 0; r < 8; ++r) {
      int mt   = r + 8 * (l >> 4);
      int mg   = m0 + wave * 16 + mt;
      int ncol = ncol0 + nt * 16 + (l & 15);
      float v  = acc[nt][r];
      int bb = mg >> 9, ii = mg & (Nc - 1);
      int h = ncol >> 6, d = ncol & (Dc - 1);
      if (wsel == 0) {          // Q: +bias, fold 1/sqrt(D)
        float q = (v + bq[ncol]) * 0.125f;
        qh[(((long)bb * Hc + h) * Nc + ii) * Dc + d] = (half_t)q;
      } else if (wsel == 1) {   // K stored transposed (b,h,d,j)
        khT[(((long)bb * Hc + h) * Dc + d) * Nc + ii] = (half_t)v;
      } else if (wsel == 2) {   // V (b,h,j,d)
        vh[(((long)bb * Hc + h) * Nc + ii) * Dc + d] = (half_t)v;
      } else {                  // gate: sigmoid, fp32
        gsig[(long)mg * CS + ncol] = 1.0f / (1.0f + __expf(-v));
      }
    }
  }
}

// ---------------------------------------------------------------------------
// Kernel 2: pair bias.  biasT[b][i][h][j] = LN(z[b,i,j,:]) @ wz.
// One wave per z-row; reads z (268 MB) exactly once.
// ---------------------------------------------------------------------------
__global__ __launch_bounds__(256) void bias_kernel(
    const float* __restrict__ z, const float* __restrict__ ln_g,
    const float* __restrict__ ln_b, const float* __restrict__ wz,
    float* __restrict__ biasT) {
  __shared__ float wzs[CZ * Hc];
  __shared__ float gs[CZ], bs[CZ];
  __shared__ float xb[8][CZ];

  const int tid = threadIdx.x;
  for (int idx = tid; idx < CZ * Hc; idx += 256) wzs[idx] = wz[idx];
  if (tid < CZ) { gs[tid] = ln_g[tid]; bs[tid] = ln_b[tid]; }
  __syncthreads();

  const int wave = tid >> 5, l = tid & 31;
  const long R = (long)blockIdx.x * 8 + wave;   // row over B*N*N
  const float* zr = z + R * CZ;

  float4 x = *(const float4*)(zr + l * 4);
  float sum = x.x + x.y + x.z + x.w;
  float sq  = x.x * x.x + x.y * x.y + x.z * x.z + x.w * x.w;
#pragma unroll
  for (int o = 16; o > 0; o >>= 1) {
    sum += __shfl_xor(sum, o, 32);
    sq  += __shfl_xor(sq, o, 32);
  }
  float mu  = sum * (1.0f / CZ);
  float var = sq * (1.0f / CZ) - mu * mu;
  float rs  = rsqrtf(var + 1e-5f);

  float* xw = xb[wave];
  int c4 = l * 4;
  xw[c4 + 0] = (x.x - mu) * rs * gs[c4 + 0] + bs[c4 + 0];
  xw[c4 + 1] = (x.y - mu) * rs * gs[c4 + 1] + bs[c4 + 1];
  xw[c4 + 2] = (x.z - mu) * rs * gs[c4 + 2] + bs[c4 + 2];
  xw[c4 + 3] = (x.w - mu) * rs * gs[c4 + 3] + bs[c4 + 3];
  __syncthreads();

  int h = l & 15, c0 = (l >> 4) * 64;
  float t = 0.0f;
  for (int c = 0; c < 64; ++c) t += xw[c0 + c] * wzs[(c0 + c) * Hc + h];
  t += __shfl_xor(t, 16, 32);
  if (l < 16) {
    long rem = R;
    int bb = (int)(rem / ((long)Nc * Nc));
    rem -= (long)bb * Nc * Nc;
    int i = (int)(rem / Nc), j = (int)(rem % Nc);
    biasT[(((long)bb * Nc + i) * Hc + h) * Nc + j] = t;
  }
}

// ---------------------------------------------------------------------------
// Kernel 3: attention.  Grid (B*H, N/32), block = 2 waves; each wave owns a
// 16x512 fp32 score strip in LDS (whole-row two-pass softmax, 320KB LDS).
// ---------------------------------------------------------------------------
__global__ __launch_bounds__(64) void attn_kernel(
    const half_t* __restrict__ qh, const half_t* __restrict__ khT,
    const half_t* __restrict__ vh, const float* __restrict__ biasT,
    const float* __restrict__ mask, const float* __restrict__ gsig,
    half_t* __restrict__ og) {
  __shared__ __align__(32) float S[2][16][Nc];   // 64 KB

  const int wave = threadIdx.x >> 5, l = threadIdx.x & 31;
  const int bh = blockIdx.x;
  const int bb = bh >> 4, h = bh & 15;
  const int i0 = blockIdx.y * 32 + wave * 16;

  const half_t* qbase = qh  + (((long)bb * Hc + h) * Nc) * Dc;  // [i][d]
  const half_t* ktb   = khT + (((long)bb * Hc + h) * Dc) * Nc;  // [d][j]
  const half_t* vb    = vh  + (((long)bb * Hc + h) * Nc) * Dc;  // [j][d]

  // Q A-fragments (K-dim = 64 -> two K32 fragments), reused for all j-tiles.
  v16h aq0, aq1;
  {
    int m = l & 15;
    const half_t* qrow = qbase + (i0 + m) * Dc;
#pragma unroll
    for (int e = 0; e < 16; ++e) {
      int k = a_frag_k(l, e);
      aq0[e] = qrow[k];
      aq1[e] = qrow[32 + k];
    }
  }
  float (*Sw)[Nc] = S[wave];

  // ---- Phase 1: S = Q K^T (pre-scaled) + bias + mask ----
  for (int j0 = 0; j0 < Nc; j0 += 16) {
    // Prefetch the next j-tile of the 33MB bias stream (global_prefetch_b8).
    if (j0 + 16 < Nc) {
      int m = l & 15;
      __builtin_prefetch(
          biasT + (((long)bb * Nc + (i0 + m)) * Hc + h) * Nc + j0 + 16, 0, 0);
    }
    v16h b0, b1;  // j = lane&15, d = a_frag_k / 32+a_frag_k
#pragma unroll
    for (int e = 0; e < 16; ++e) {
      int kk = a_frag_k(l, e);
      b0[e] = ktb[(long)kk * Nc + j0 + (l & 15)];
      b1[e] = ktb[(long)(32 + kk) * Nc + j0 + (l & 15)];
    }
    v8f acc = {};
    acc = wmma_f16(aq0, b0, acc);
    acc = wmma_f16(aq1, b1, acc);
    int n = l & 15, jg = j0 + n;
    float mb = (1.0f - mask[bb * Nc + jg]) * (-1.0e6f);
#pragma unroll
    for (int r = 0; r < 8; ++r) {
      int m = r + 8 * (l >> 4);
      float bv = biasT[(((long)bb * Nc + (i0 + m)) * Hc + h) * Nc + jg];
      Sw[m][jg] = acc[r] + bv + mb;
    }
  }

  // ---- Phase 2: softmax over j (lane pair (l, l+16) splits row l&15) ----
  {
    int m = l & 15, c0 = (l >> 4) * (Nc / 2);
    float mx = -3.0e38f;
    for (int c = 0; c < Nc / 2; ++c) mx = fmaxf(mx, Sw[m][c0 + c]);
    mx = fmaxf(mx, __shfl_xor(mx, 16, 32));
    float sum = 0.0f;
    for (int c = 0; c < Nc / 2; ++c) {
      float e = __expf(Sw[m][c0 + c] - mx);
      Sw[m][c0 + c] = e;
      sum += e;
    }
    sum += __shfl_xor(sum, 16, 32);
    float inv = 1.0f / sum;
    for (int c = 0; c < Nc / 2; ++c) Sw[m][c0 + c] *= inv;
  }

  // ---- Phase 3: O = P @ V, gate, store f16 for the output GEMM ----
#pragma unroll
  for (int d0 = 0; d0 < Dc; d0 += 16) {
    v8f acc = {};
    int m = l & 15;
    for (int j0 = 0; j0 < Nc; j0 += 32) {
      v16h a;
#pragma unroll
      for (int e = 0; e < 16; ++e)
        a[e] = (half_t)Sw[m][j0 + a_frag_k(l, e)];
      v16h bf;  // d = lane&15, j = a_frag_k
#pragma unroll
      for (int e = 0; e < 16; ++e)
        bf[e] = vb[(long)(j0 + a_frag_k(l, e)) * Dc + d0 + (l & 15)];
      acc = wmma_f16(a, bf, acc);
    }
    int n = l & 15;
#pragma unroll
    for (int r = 0; r < 8; ++r) {
      int mt = r + 8 * (l >> 4);
      long row = (long)bb * Nc + (i0 + mt);
      int c = h * Dc + d0 + n;
      float g = gsig[row * CS + c];
      og[row * CS + c] = (half_t)(g * acc[r]);
    }
  }
}

// ---------------------------------------------------------------------------
// Kernel 4: out = (g*o) @ wo.   og is f16 and is staged to LDS byte-for-byte
// with GLOBAL_LOAD_ASYNC_TO_LDS_B128 (ASYNCcnt path); wo is f32 -> f16 on the
// VALU path; accumulate f32.
// ---------------------------------------------------------------------------
__global__ __launch_bounds__(128) void out_gemm_kernel(
    const half_t* __restrict__ og, const float* __restrict__ wo,
    float* __restrict__ out) {
  __shared__ __align__(32) half_t As[64 * 32];
  __shared__ __align__(32) half_t Bs[32 * 64];

  const int m0 = blockIdx.x * 64, n0 = blockIdx.y * 64;
  const int tid = threadIdx.x, wave = tid >> 5, l = tid & 31;

  v8f acc[4];
#pragma unroll
  for (int nt = 0; nt < 4; ++nt) { v8f z = {}; acc[nt] = z; }

  const int r_a  = tid >> 1, ch_a = (tid & 1) * 16;        // A staging slice
  const unsigned lds_a = (unsigned)(uintptr_t)(As + r_a * 32 + ch_a);

  for (int k0 = 0; k0 < CS; k0 += 32) {
    // A tile: async global->LDS copy (no conversion needed, og is f16)
    async_copy_b256(lds_a, og + (long)(m0 + r_a) * CS + k0 + ch_a);
    // B tile: f32 -> f16 via VALU
    {
      int kr = tid >> 2, cq = (tid & 3) * 16;
      const float* src = wo + (long)(k0 + kr) * CS + n0 + cq;
#pragma unroll
      for (int u = 0; u < 16; ++u) Bs[kr * 64 + cq + u] = (half_t)src[u];
    }
    wait_asynccnt0();   // drain this wave's async copies before the barrier
    __syncthreads();
    v16h a = load_a_f16(As + (wave * 16) * 32, 32);
#pragma unroll
    for (int nt = 0; nt < 4; ++nt) {
      v16h b;
#pragma unroll
      for (int e = 0; e < 16; ++e) b[e] = Bs[a_frag_k(l, e) * 64 + nt * 16 + (l & 15)];
      acc[nt] = wmma_f16(a, b, acc[nt]);
    }
    __syncthreads();
  }

#pragma unroll
  for (int nt = 0; nt < 4; ++nt) {
#pragma unroll
    for (int r = 0; r < 8; ++r) {
      int mg = m0 + wave * 16 + r + 8 * (l >> 4);
      int ng = n0 + nt * 16 + (l & 15);
      out[(long)mg * CS + ng] = acc[nt][r];
    }
  }
}

// ---------------------------------------------------------------------------
extern "C" void kernel_launch(void* const* d_in, const int* in_sizes, int n_in,
                              void* d_out, int out_size, void* d_ws,
                              size_t ws_size, hipStream_t stream) {
  const float* s    = (const float*)d_in[0];
  const float* z    = (const float*)d_in[1];
  const float* mask = (const float*)d_in[2];
  const float* k_in = (const float*)d_in[3];
  const float* wq   = (const float*)d_in[4];
  const float* bq   = (const float*)d_in[5];
  const float* wk   = (const float*)d_in[6];
  const float* wv   = (const float*)d_in[7];
  const float* wg   = (const float*)d_in[8];
  const float* ln_g = (const float*)d_in[9];
  const float* ln_b = (const float*)d_in[10];
  const float* wz   = (const float*)d_in[11];
  const float* wo   = (const float*)d_in[12];
  // d_in[13] = multiplicity (== 1 in this configuration; bias repeat is identity)
  (void)in_sizes; (void)n_in; (void)out_size; (void)ws_size;

  char* ws = (char*)d_ws;
  const size_t MiB = 1u << 20;
  half_t* qh    = (half_t*)(ws + 0 * MiB);   // B*H*N*D f16  (2 MiB)
  half_t* khT   = (half_t*)(ws + 2 * MiB);   // B*H*D*N f16  (2 MiB)
  half_t* vh    = (half_t*)(ws + 4 * MiB);   // B*H*N*D f16  (2 MiB)
  float*  gsig  = (float*)(ws + 6 * MiB);    // B*N*CS  f32  (4 MiB)
  float*  biasT = (float*)(ws + 10 * MiB);   // B*N*H*N f32  (32 MiB)
  half_t* og    = (half_t*)(ws + 42 * MiB);  // B*N*CS  f16  (2 MiB)

  // 1) projections q/k/v/gate (WMMA GEMM over concatenated weights)
  proj_kernel<<<dim3(BN / 64, (4 * CS) / 64), 128, 0, stream>>>(
      s, k_in, wq, bq, wk, wv, wg, qh, khT, vh, gsig);

  // 2) pair bias: LN(z) @ wz -> biasT (reads z once)
  bias_kernel<<<(Bc * Nc * Nc) / 8, 256, 0, stream>>>(z, ln_g, ln_b, wz,
                                                      biasT);

  // 3) attention with whole-row LDS softmax
  attn_kernel<<<dim3(Bc * Hc, Nc / 32), 64, 0, stream>>>(
      qh, khT, vh, biasT, mask, gsig, og);

  // 4) output projection
  out_gemm_kernel<<<dim3(BN / 64, CS / 64), 128, 0, stream>>>(og, wo,
                                                              (float*)d_out);
}
